// GCNModel_33818572488717
// MI455X (gfx1250) — compile-verified, matrix operand order checked by reading, not run on hardware
//
#include <hip/hip_runtime.h>
#include <hip/hip_bf16.h>

typedef __bf16 bf16;
typedef __attribute__((ext_vector_type(8)))  __bf16 v8bf;
typedef __attribute__((ext_vector_type(16))) __bf16 v16bf;
typedef __attribute__((ext_vector_type(8)))  float  v8f;

// ---------------------------------------------------------------- utilities
__global__ void fill_f32(float* __restrict__ p, int n, float v) {
    int i = blockIdx.x * blockDim.x + threadIdx.x;
    if (i < n) p[i] = v;
}

// vectorized fill: n4 = count of float4
__global__ void fill_f32x4(float4* __restrict__ p, int n4, float v) {
    int i = blockIdx.x * blockDim.x + threadIdx.x;
    if (i < n4) p[i] = make_float4(v, v, v, v);
}

// deg[i] starts at 1.0 (self loop); count in-edges on dst
__global__ void deg_count(const int* __restrict__ dst, float* __restrict__ deg, int E) {
    int e = blockIdx.x * blockDim.x + threadIdx.x;
    if (e < E) atomicAdd(&deg[dst[e]], 1.0f);
}

__global__ void deg_rsqrt(const float* __restrict__ deg, float* __restrict__ dis, int N) {
    int i = blockIdx.x * blockDim.x + threadIdx.x;
    if (i < N) dis[i] = rsqrtf(deg[i]);
}

__global__ void edge_norm(const int* __restrict__ src, const int* __restrict__ dst,
                          const float* __restrict__ dis, float* __restrict__ norm, int E) {
    int e = blockIdx.x * blockDim.x + threadIdx.x;
    if (e < E) norm[e] = dis[src[e]] * dis[dst[e]];
}

// ------------------------------------------------------ WMMA GEMM  T = A(Mx64) * W(64x64)
// block = 256 threads = 8 waves; each wave does a 16-row strip (block covers 128 rows).
// A & W^T staged into LDS as bf16; fragments built with 16B ds loads matching the
// CDNA5 16-bit A/B VGPR layout (elements 0..7 -> K = k0+8h.., 8..15 -> K = k0+16+8h..).
__device__ inline v16bf load_frag16(const bf16* __restrict__ rowPtr, int k0, int half) {
    v8bf lo = *(const v8bf*)(rowPtr + k0 + 8 * half);
    v8bf hi = *(const v8bf*)(rowPtr + k0 + 16 + 8 * half);
    return __builtin_shufflevector(lo, hi, 0,1,2,3,4,5,6,7,8,9,10,11,12,13,14,15);
}

__global__ __launch_bounds__(256)
void gemm64_wmma_bf16(const float* __restrict__ A, const float* __restrict__ W,
                      float* __restrict__ T, int M) {
    __shared__ alignas(16) bf16 sA[128][72];   // 128 rows x 64 cols (+8 pad)
    __shared__ alignas(16) bf16 sW[64][72];    // sW[n][k] = W[k][n]  (transposed)

    const int tid = threadIdx.x;
    const int rowBase = blockIdx.x * 128;

    // Stage W (64x64 f32, coalesced) -> LDS transposed bf16
#pragma unroll
    for (int i = 0; i < 16; ++i) {
        int idx = tid + i * 256;           // 4096 elements
        int k = idx >> 6, n = idx & 63;
        sW[n][k] = (bf16)W[idx];
    }
    // Stage 128 rows of A as float4 (global_load_b128, coalesced), zero-padded tail
#pragma unroll
    for (int i = 0; i < 8; ++i) {
        int idx = tid + i * 256;           // 2048 float4 = 8192 floats
        int r = idx >> 4, q = idx & 15;    // q = float4 index within the row
        int g = rowBase + r;
        float4 v = make_float4(0.f, 0.f, 0.f, 0.f);
        if (g < M) v = ((const float4*)A)[(size_t)g * 16 + q];
        bf16* d = &sA[r][q * 4];
        d[0] = (bf16)v.x; d[1] = (bf16)v.y; d[2] = (bf16)v.z; d[3] = (bf16)v.w;
    }
    __syncthreads();

    const int wave = tid >> 5, lane = tid & 31;
    const int r0   = wave * 16;
    const int mrow = lane & 15, half = lane >> 4;

    // A fragments for K 0..31 and 32..63
    const bf16* aRow = &sA[r0 + mrow][0];
    v16bf a0 = load_frag16(aRow, 0, half);
    v16bf a1 = load_frag16(aRow, 32, half);

    const int gr = rowBase + r0;
    const bool fullTile = (gr + 16 <= M);

#pragma unroll
    for (int j = 0; j < 4; ++j) {          // four 16-col output tiles
        int n0 = j * 16;
        const bf16* bRow = &sW[n0 + mrow][0];
        v16bf b0 = load_frag16(bRow, 0, half);
        v16bf b1 = load_frag16(bRow, 32, half);
        v8f acc = {};
        acc = __builtin_amdgcn_wmma_f32_16x16x32_bf16(false, a0, false, b0,
                                                      (short)0, acc, false, false);
        acc = __builtin_amdgcn_wmma_f32_16x16x32_bf16(false, a1, false, b1,
                                                      (short)0, acc, false, false);
        // C/D layout: VGPR r -> row gr+r+8*half, col = n0 + mrow.
        // Single base pointer; rows stride 64 floats -> immediate-offset stores.
        float* p = T + (size_t)(gr + 8 * half) * 64 + n0 + mrow;
        if (fullTile) {
#pragma unroll
            for (int r = 0; r < 8; ++r) p[(size_t)r * 64] = acc[r];
        } else {
#pragma unroll
            for (int r = 0; r < 8; ++r)
                if (gr + r + 8 * half < M) p[(size_t)r * 64] = acc[r];
        }
    }
}

// ---------------------------------------------------- edge scatter (64 threads / edge)
__global__ void scatter_edges(const float* __restrict__ t, const int* __restrict__ src,
                              const int* __restrict__ dst, const float* __restrict__ norm,
                              float* __restrict__ accum, int E) {
    long long idx = (long long)blockIdx.x * blockDim.x + threadIdx.x;
    int e = (int)(idx >> 6);
    int c = (int)(idx & 63);
    if (e < E) {
        float v = t[(size_t)src[e] * 64 + c] * norm[e];
        atomicAdd(&accum[(size_t)dst[e] * 64 + c], v);
    }
}

// self-loop contribution + bias + ReLU, float4 per thread (16 threads / node)
__global__ void finalize_relu(const float4* __restrict__ accum, const float4* __restrict__ t,
                              const float* __restrict__ dis, const float4* __restrict__ b,
                              float4* __restrict__ h, int N) {
    long long idx = (long long)blockIdx.x * blockDim.x + threadIdx.x;
    int i = (int)(idx >> 4);
    int q = (int)(idx & 15);
    if (i < N) {
        float d2 = dis[i]; d2 *= d2;
        float4 a = accum[(size_t)i * 16 + q];
        float4 s = t[(size_t)i * 16 + q];
        float4 bb = b[q];
        float4 r;
        r.x = fmaxf(a.x + s.x * d2 + bb.x, 0.0f);
        r.y = fmaxf(a.y + s.y * d2 + bb.y, 0.0f);
        r.z = fmaxf(a.z + s.z * d2 + bb.z, 0.0f);
        r.w = fmaxf(a.w + s.w * d2 + bb.w, 0.0f);
        h[(size_t)i * 16 + q] = r;
    }
}

// ---------------------------------------------------- mean pool over batch ids
__global__ void pool_accum(const float* __restrict__ h, const int* __restrict__ batch,
                           float* __restrict__ sums, float* __restrict__ cnts, int N) {
    long long idx = (long long)blockIdx.x * blockDim.x + threadIdx.x;
    int i = (int)(idx >> 6);
    int c = (int)(idx & 63);
    if (i < N) {
        int g = batch[i];
        atomicAdd(&sums[(size_t)g * 64 + c], h[(size_t)i * 64 + c]);
        if (c == 0) atomicAdd(&cnts[g], 1.0f);
    }
}

// tiny head MLP: one thread per graph (G=2048; weights broadcast -> scalar-cached)
__global__ void head_mlp(const float* __restrict__ sums, const float* __restrict__ cnts,
                         const float* __restrict__ Wl1, const float* __restrict__ bl1,
                         const float* __restrict__ Wl2, const float* __restrict__ bl2,
                         float* __restrict__ out, int G) {
    int g = blockIdx.x * blockDim.x + threadIdx.x;
    if (g >= G) return;
    float inv = 1.0f / fmaxf(cnts[g], 1.0f);
    float p[64];
#pragma unroll
    for (int k = 0; k < 64; ++k) p[k] = sums[(size_t)g * 64 + k] * inv;
    float o = bl2[0];
    for (int j = 0; j < 32; ++j) {
        float a = bl1[j];
#pragma unroll
        for (int k = 0; k < 64; ++k) a += p[k] * Wl1[k * 32 + j];
        o += fmaxf(a, 0.0f) * Wl2[j];
    }
    out[g] = o;
}

// ---------------------------------------------------------------- launcher
extern "C" void kernel_launch(void* const* d_in, const int* in_sizes, int n_in,
                              void* d_out, int out_size, void* d_ws, size_t ws_size,
                              hipStream_t stream) {
    const float* x    = (const float*)d_in[0];
    const int*   ei   = (const int*)  d_in[1];   // [2, E] (src row, dst row)
    const int*   bat  = (const int*)  d_in[2];   // [N]
    const float* W1   = (const float*)d_in[3];
    const float* b1   = (const float*)d_in[4];
    const float* W2   = (const float*)d_in[5];
    const float* b2   = (const float*)d_in[6];
    const float* Wl1  = (const float*)d_in[7];
    const float* bl1  = (const float*)d_in[8];
    const float* Wl2  = (const float*)d_in[9];
    const float* bl2  = (const float*)d_in[10];
    float* out = (float*)d_out;

    const int N = in_sizes[0] / 64;
    const int E = in_sizes[1] / 2;
    const int G = out_size;
    const int* src = ei;
    const int* dst = ei + E;

    // workspace carve-out (256B aligned)
    char* ws = (char*)d_ws;
    size_t off = 0;
    auto carve = [&](size_t bytes) {
        void* p = ws + off;
        off = (off + bytes + 255) & ~(size_t)255;
        return p;
    };
    float* t     = (float*)carve((size_t)N * 64 * 4);
    float* accum = (float*)carve((size_t)N * 64 * 4);
    float* h     = (float*)carve((size_t)N * 64 * 4);
    float* deg   = (float*)carve((size_t)N * 4);
    float* dis   = (float*)carve((size_t)N * 4);
    float* norm  = (float*)carve((size_t)E * 4);
    float* sums  = (float*)carve((size_t)G * 64 * 4);
    float* cnts  = (float*)carve((size_t)G * 4);
    (void)ws_size; (void)n_in;

    const int TB = 256;
    dim3 blk(TB);
    auto g1 = [&](long long n) { return dim3((unsigned)((n + TB - 1) / TB)); };

    // degrees / symmetric norm
    fill_f32<<<g1(N), blk, 0, stream>>>(deg, N, 1.0f);          // self-loop
    deg_count<<<g1(E), blk, 0, stream>>>(dst, deg, E);
    deg_rsqrt<<<g1(N), blk, 0, stream>>>(deg, dis, N);
    edge_norm<<<g1(E), blk, 0, stream>>>(src, dst, dis, norm, E);

    const dim3 gemmGrid((N + 127) / 128);
    const long long NC  = (long long)N * 64;      // node-channel elements
    const long long NC4 = NC / 4;                 // float4 count
    const long long EC  = (long long)E * 64;

    // ---- layer 1
    gemm64_wmma_bf16<<<gemmGrid, blk, 0, stream>>>(x, W1, t, N);
    fill_f32x4<<<g1(NC4), blk, 0, stream>>>((float4*)accum, (int)NC4, 0.0f);
    scatter_edges<<<g1(EC), blk, 0, stream>>>(t, src, dst, norm, accum, E);
    finalize_relu<<<g1(NC / 4), blk, 0, stream>>>((const float4*)accum, (const float4*)t,
                                                  dis, (const float4*)b1, (float4*)h, N);

    // ---- layer 2
    gemm64_wmma_bf16<<<gemmGrid, blk, 0, stream>>>(h, W2, t, N);
    fill_f32x4<<<g1(NC4), blk, 0, stream>>>((float4*)accum, (int)NC4, 0.0f);
    scatter_edges<<<g1(EC), blk, 0, stream>>>(t, src, dst, norm, accum, E);
    finalize_relu<<<g1(NC / 4), blk, 0, stream>>>((const float4*)accum, (const float4*)t,
                                                  dis, (const float4*)b2, (float4*)h, N);

    // ---- mean pool + head
    fill_f32x4<<<g1((long long)G * 16), blk, 0, stream>>>((float4*)sums, G * 16, 0.0f);
    fill_f32<<<g1(G), blk, 0, stream>>>(cnts, G, 0.0f);
    pool_accum<<<g1(NC), blk, 0, stream>>>(h, bat, sums, cnts, N);
    head_mlp<<<g1(G), blk, 0, stream>>>(sums, cnts, Wl1, bl1, Wl2, bl2, out, G);
}
